// Net_12541304505101
// MI455X (gfx1250) — matrix operand order repaired and checked by reading, not compiled, not run on hardware
//
#include <hip/hip_runtime.h>
#include <hip/hip_bf16.h>

// ---------------------------------------------------------------------------
// Problem constants
// ---------------------------------------------------------------------------
#define BATCH 4096
#define NEXP  8
#define DDIM  9216      // 64 * 12 * 12
#define HDIM  128
#define ODIM  10

// conv geometry
#define C1OUT 32
#define C1HW  26        // 28 - 2
#define C2OUT 64
#define C2HW  24        // 26 - 2
#define POOLHW 12

#define K2    288       // 32 ic * 9 taps
#define NTOK2 (BATCH * C2HW * C2HW)   // conv2 GEMM M = 2,359,296
#define KT1   288       // DDIM / 32
#define WC2ELEMS (9 * C2OUT * C1OUT)  // 18432 bf16 = 36864 bytes = 9216 dwords

// ---------------------------------------------------------------------------
// CDNA5 WMMA / descriptor types
// ---------------------------------------------------------------------------
typedef __attribute__((ext_vector_type(16))) __bf16        v16bf;
typedef __attribute__((ext_vector_type(8)))  float         v8f;
typedef __attribute__((ext_vector_type(4)))  unsigned int  u32x4;
typedef __attribute__((ext_vector_type(4)))  float         f32x4;
typedef __attribute__((ext_vector_type(8)))  int           i32x8;
typedef __attribute__((ext_vector_type(4)))  int           i32x4;

union FragAB { v16bf v; u32x4 q[2]; };

__device__ __forceinline__ unsigned short f2bf(float f) {
    unsigned int u = __float_as_uint(f);
    unsigned int r = (u + 0x7FFFu + ((u >> 16) & 1u)) >> 16;
    return (unsigned short)r;
}
__device__ __forceinline__ float bf2f(unsigned short b) {
    return __uint_as_float(((unsigned int)b) << 16);
}

// ---------------------------------------------------------------------------
// 0) zero per-expert counters
// ---------------------------------------------------------------------------
__global__ void zero_cnt_kernel(int* cnt) {
    if (threadIdx.x < NEXP) cnt[threadIdx.x] = 0;
}

// ---------------------------------------------------------------------------
// 1) conv1 + relu -> h1 [B][26][26][32] bf16 (NHWC)
// ---------------------------------------------------------------------------
__global__ __launch_bounds__(256)
void conv1_kernel(const float* __restrict__ x, const float* __restrict__ w,
                  const float* __restrict__ bias, unsigned short* __restrict__ h1) {
    int t = blockIdx.x * 256 + threadIdx.x;            // B*26*26
    if (t >= BATCH * C1HW * C1HW) return;
    int b  = t / (C1HW * C1HW);
    int s  = t - b * (C1HW * C1HW);
    int y  = s / C1HW;
    int xp = s - y * C1HW;

    const float* xin = x + ((long)b * 28 + y) * 28 + xp;
    float patch[9];
#pragma unroll
    for (int ky = 0; ky < 3; ky++)
#pragma unroll
        for (int kx = 0; kx < 3; kx++)
            patch[ky * 3 + kx] = xin[ky * 28 + kx];

    unsigned int packed[16];
#pragma unroll
    for (int pc = 0; pc < 16; pc++) {
        unsigned int pv = 0;
#pragma unroll
        for (int half = 0; half < 2; half++) {
            int oc = pc * 2 + half;
            float acc = bias[oc];
#pragma unroll
            for (int j = 0; j < 9; j++) acc = fmaf(patch[j], w[oc * 9 + j], acc);
            acc = fmaxf(acc, 0.0f);
            pv |= ((unsigned int)f2bf(acc)) << (16 * half);
        }
        packed[pc] = pv;
    }
    u32x4* dst = (u32x4*)(h1 + (long)t * 32);
#pragma unroll
    for (int q = 0; q < 4; q++) {
        u32x4 v = { packed[4*q], packed[4*q+1], packed[4*q+2], packed[4*q+3] };
        dst[q] = v;
    }
}

// ---------------------------------------------------------------------------
// 2a) conv2 weights [64][32][3][3] f32 -> [tap=9][oc=64][ic=32] bf16
// ---------------------------------------------------------------------------
__global__ void conv2w_convert_kernel(const float* __restrict__ w,
                                      unsigned short* __restrict__ wb) {
    int i = blockIdx.x * 256 + threadIdx.x;            // 9*64*32
    if (i >= WC2ELEMS) return;
    int ic = i & 31;
    int oc = (i >> 5) & 63;
    int j  = i >> 11;                                  // tap
    wb[i] = f2bf(w[(oc * 32 + ic) * 9 + j]);
}

// ---------------------------------------------------------------------------
// 2b) w1 [E][D][H] f32 -> [e][kt=288][h=128][kin=32] bf16
// ---------------------------------------------------------------------------
__global__ void w1_convert_kernel(const float* __restrict__ w1,
                                  unsigned short* __restrict__ w1b) {
    long i = (long)blockIdx.x * 256 + threadIdx.x;     // 8*288*128*32
    if (i >= (long)NEXP * KT1 * HDIM * 32) return;
    int  kin = (int)(i & 31);
    long r   = i >> 5;
    int  h   = (int)(r & 127);
    r >>= 7;
    int  kt  = (int)(r % KT1);
    int  e   = (int)(r / KT1);
    int  d   = kt * 32 + kin;
    w1b[i] = f2bf(w1[((long)e * DDIM + d) * HDIM + h]);
}

// ---------------------------------------------------------------------------
// 3) conv2 implicit GEMM (WMMA bf16) + bias + relu + fused 2x2 maxpool
//    Weights (36 KB) staged to LDS once per workgroup with the Tensor Data
//    Mover (TENSORcnt), then B fragments served via ds_load_b128.
//    token order: t = b*576 + (yp*12+xp)*4 + (dy*2+dx)  (pool members adjacent)
//    block = 256 thr = 8 waves; wave = 16 tokens (4 pool windows) x 64 oc
//    output h[b][d] bf16 with d = c*144 + yp*12 + xp  (matches NCHW flatten)
// ---------------------------------------------------------------------------
__global__ __launch_bounds__(256)
void conv2_kernel(const unsigned short* __restrict__ h1,
                  const unsigned short* __restrict__ wb,
                  const float* __restrict__ bias,
                  unsigned short* __restrict__ h) {
    __shared__ unsigned short wS[WC2ELEMS];            // 36,864 bytes

    const int tid  = threadIdx.x;
    const int lane = tid & 31;
    const int wav  = tid >> 5;

#if __has_builtin(__builtin_amdgcn_tensor_load_to_lds)
    if (wav == 0) {
        // Tensor DMA descriptor (D#): 1-D tensor, 9216 x 4-byte elements.
        const unsigned long long ga = (unsigned long long)wb;
        // LDS byte offset = low 32 bits of the generic shared-aperture address
        const unsigned int lds_addr = (unsigned int)(unsigned long long)(const void*)wS;

        u32x4 g0;
        g0.x = 1u;                                        // count=1 (valid user D#)
        g0.y = lds_addr;                                  // lds_addr [63:32]
        g0.z = (unsigned int)(ga & 0xffffffffull);        // global_addr [95:64]
        g0.w = (unsigned int)((ga >> 32) & 0x01ffffffull) // global_addr [120:96]
             | 0x80000000u;                               // type=2 ("image") [127:126]

        i32x8 g1;
        g1[0] = 0x00020000;          // wg_mask=0, data_size=2 (4B), no iterate/pad
        g1[1] = 0x24000000;          // tensor_dim0 = 9216  (low 16 bits @ [31:16])
        g1[2] = 0x00010000;          // tensor_dim0 hi = 0 ; tensor_dim1 = 1
        g1[3] = 0x24000000;          // tensor_dim1 hi = 0 ; tile_dim0 = 9216
        g1[4] = 0x00000001;          // tile_dim1 = 1 ; tile_dim2 = 0
        g1[5] = 9216;                // tensor_dim0_stride lo
        g1[6] = 0x24000000;          // stride0 hi = 0 ; tensor_dim1_stride lo = 9216
        g1[7] = 0;                   // tensor_dim1_stride hi

        i32x4 z4 = {0, 0, 0, 0};
#if __clang_major__ >= 23
        i32x8 z8 = {0, 0, 0, 0, 0, 0, 0, 0};
        __builtin_amdgcn_tensor_load_to_lds(g0, g1, z4, z4, z8, 0);
#else
        __builtin_amdgcn_tensor_load_to_lds(g0, g1, z4, z4, 0);
#endif
#if __has_builtin(__builtin_amdgcn_s_wait_tensorcnt)
        __builtin_amdgcn_s_wait_tensorcnt(0);
#else
        asm volatile("s_wait_tensorcnt 0x0" ::: "memory");
#endif
    }
#else
    // Fallback: cooperative b128 copy global -> LDS
    for (int i = tid; i < WC2ELEMS / 8; i += 256)
        ((u32x4*)wS)[i] = ((const u32x4*)wb)[i];
#endif
    __syncthreads();

    const int mbase = blockIdx.x * 128 + wav * 16;     // wave's first token
    const int row   = lane & 15;
    const int t     = mbase + row;

    const int b  = t / 576;
    const int r  = t - b * 576;
    const int wi = r >> 2;                             // yp*12+xp
    const int m  = r & 3;
    const int yp = wi / POOLHW, xp = wi - yp * POOLHW;
    const int y0 = 2 * yp + (m >> 1);
    const int x0 = 2 * xp + (m & 1);

    // A fragment: lanes<16 take ic {0..15}, lanes>=16 take {16..31}
    const unsigned short* abase =
        h1 + (((long)b * C1HW + y0) * C1HW + x0) * 32 + (lane & 16);
    // B fragment (from LDS): lane column oc = n*16 + lane%16 ; K run base (lane&16)
    const unsigned short* bbase = wS + (lane & 15) * 32 + (lane & 16);

    v8f acc0 = {}, acc1 = {}, acc2 = {}, acc3 = {};

#pragma unroll
    for (int ky = 0; ky < 3; ky++) {
#pragma unroll
        for (int kx = 0; kx < 3; kx++) {
            const int j = ky * 3 + kx;
            FragAB A;
            const unsigned short* ap = abase + (ky * C1HW + kx) * 32;
            A.q[0] = *(const u32x4*)(ap);
            A.q[1] = *(const u32x4*)(ap + 8);

            const unsigned short* bp = bbase + j * (C2OUT * 32);
            FragAB B0, B1, B2, B3;
            B0.q[0] = *(const u32x4*)(bp);            B0.q[1] = *(const u32x4*)(bp + 8);
            B1.q[0] = *(const u32x4*)(bp + 512);      B1.q[1] = *(const u32x4*)(bp + 520);
            B2.q[0] = *(const u32x4*)(bp + 1024);     B2.q[1] = *(const u32x4*)(bp + 1032);
            B3.q[0] = *(const u32x4*)(bp + 1536);     B3.q[1] = *(const u32x4*)(bp + 1544);

            acc0 = __builtin_amdgcn_wmma_f32_16x16x32_bf16(false, A.v, false, B0.v, (short)0, acc0, false, false);
            acc1 = __builtin_amdgcn_wmma_f32_16x16x32_bf16(false, A.v, false, B1.v, (short)0, acc1, false, false);
            acc2 = __builtin_amdgcn_wmma_f32_16x16x32_bf16(false, A.v, false, B2.v, (short)0, acc2, false, false);
            acc3 = __builtin_amdgcn_wmma_f32_16x16x32_bf16(false, A.v, false, B3.v, (short)0, acc3, false, false);
        }
    }

    // pooled windows for this lane: lanes<16 -> windows 0,1 ; lanes>=16 -> 2,3
    const int sbase = ((mbase % 576) >> 2);            // window spatial base (yp*12+xp)
    const int s0 = sbase + ((lane >> 4) << 1);
    const int s1 = s0 + 1;
    const long outb = (long)b * DDIM;

    v8f accs[4] = { acc0, acc1, acc2, acc3 };
#pragma unroll
    for (int n = 0; n < 4; n++) {
        v8f c = accs[n];
        const int oc = n * 16 + (lane & 15);
        const float bb = bias[oc];
        float p0 = fmaxf(fmaxf(c[0], c[1]), fmaxf(c[2], c[3]));
        float p1 = fmaxf(fmaxf(c[4], c[5]), fmaxf(c[6], c[7]));
        p0 = fmaxf(p0 + bb, 0.0f);
        p1 = fmaxf(p1 + bb, 0.0f);
        h[outb + oc * 144 + s0] = f2bf(p0);
        h[outb + oc * 144 + s1] = f2bf(p1);
    }
}

// ---------------------------------------------------------------------------
// 4) gate: logits = h @ gate_w, softmax, argmax -> scale + per-expert gather
//    one wave per token
// ---------------------------------------------------------------------------
__global__ __launch_bounds__(256)
void gate_kernel(const unsigned short* __restrict__ h,
                 const float* __restrict__ gw,
                 float* __restrict__ scale,
                 int* __restrict__ cnt,
                 int* __restrict__ toklist) {
    const int lane = threadIdx.x & 31;
    const int tok  = (blockIdx.x * blockDim.x + threadIdx.x) >> 5;
    if (tok >= BATCH) return;

    const unsigned short* hp = h + (long)tok * DDIM;
    float acc[NEXP] = {};
    for (int k = lane; k < DDIM; k += 32) {
        const float hv = bf2f(hp[k]);
        const f32x4 g0 = *(const f32x4*)(gw + (long)k * NEXP);
        const f32x4 g1 = *(const f32x4*)(gw + (long)k * NEXP + 4);
#pragma unroll
        for (int e = 0; e < 4; e++) acc[e]     = fmaf(hv, g0[e], acc[e]);
#pragma unroll
        for (int e = 0; e < 4; e++) acc[e + 4] = fmaf(hv, g1[e], acc[e + 4]);
    }
#pragma unroll
    for (int off = 16; off > 0; off >>= 1)
#pragma unroll
        for (int e = 0; e < NEXP; e++)
            acc[e] += __shfl_xor(acc[e], off, 32);

    if (lane == 0) {
        float mx = acc[0]; int best = 0;
#pragma unroll
        for (int e = 1; e < NEXP; e++)
            if (acc[e] > mx) { mx = acc[e]; best = e; }   // first-max like argmax
        float s = 0.0f;
#pragma unroll
        for (int e = 0; e < NEXP; e++) s += __expf(acc[e] - mx);
        scale[tok] = 1.0f / s;                            // exp(mx-mx)/s
        const int slot = atomicAdd(&cnt[best], 1);
        toklist[best * BATCH + slot] = tok;
    }
}

// ---------------------------------------------------------------------------
// 5) sparse expert FFN: for each expert, M-tiles of 16 gathered tokens.
//    layer1: 16x128 <- K=9216 via WMMA bf16 ; bias+relu in LDS
//    layer2: 128x10 dense + scale + log_softmax
//    block = 128 thr = 4 waves; wave owns 2 N-tiles (32 cols of H)
// ---------------------------------------------------------------------------
__global__ __launch_bounds__(128)
void expert_kernel(const unsigned short* __restrict__ h,
                   const unsigned short* __restrict__ w1b,
                   const float* __restrict__ b1,
                   const float* __restrict__ w2,
                   const float* __restrict__ b2,
                   const float* __restrict__ scale,
                   const int* __restrict__ cnt,
                   const int* __restrict__ toklist,
                   float* __restrict__ out) {
    const int e  = blockIdx.y;
    const int mt = blockIdx.x;
    const int count = cnt[e];
    if (mt * 16 >= count) return;

    __shared__ float heS[16][HDIM + 4];
    __shared__ float oeS[16][ODIM];
    __shared__ int   tokS[16];

    const int tid  = threadIdx.x;
    const int lane = tid & 31;
    const int wav  = tid >> 5;

    if (tid < 16) {
        int i = mt * 16 + tid;
        tokS[tid] = toklist[e * BATCH + ((i < count) ? i : (count - 1))];
    }
    __syncthreads();

    const unsigned short* ap0 =
        h + (long)tokS[lane & 15] * DDIM + (lane & 16);
    const unsigned short* bp0 =
        w1b + (long)e * KT1 * HDIM * 32
            + (wav * 32 + (lane & 15)) * 32 + (lane & 16);

    v8f a0 = {}, a1 = {};
#pragma unroll 4
    for (int kt = 0; kt < KT1; kt++) {
        FragAB A, B0, B1;
        const unsigned short* ap = ap0 + kt * 32;
        A.q[0] = *(const u32x4*)(ap);
        A.q[1] = *(const u32x4*)(ap + 8);
        const unsigned short* bq = bp0 + (long)kt * (HDIM * 32);
        B0.q[0] = *(const u32x4*)(bq);        B0.q[1] = *(const u32x4*)(bq + 8);
        B1.q[0] = *(const u32x4*)(bq + 512);  B1.q[1] = *(const u32x4*)(bq + 520);
        a0 = __builtin_amdgcn_wmma_f32_16x16x32_bf16(false, A.v, false, B0.v, (short)0, a0, false, false);
        a1 = __builtin_amdgcn_wmma_f32_16x16x32_bf16(false, A.v, false, B1.v, (short)0, a1, false, false);
    }

    const int rbase = (lane >> 4) * 8;
#pragma unroll
    for (int n = 0; n < 2; n++) {
        v8f c = n ? a1 : a0;
        const int col = wav * 32 + n * 16 + (lane & 15);
        const float bb = b1[e * HDIM + col];
#pragma unroll
        for (int rr = 0; rr < 8; rr++)
            heS[rbase + rr][col] = fmaxf(c[rr] + bb, 0.0f);
    }
    __syncthreads();

    // layer 2: oe[t][o] = he[t][:] . w2[e][:,o] + b2 ; then * gate scale
    for (int i = tid; i < 16 * ODIM; i += 128) {
        const int t = i & 15, o = i >> 4;
        float s = b2[e * ODIM + o];
        const float* w2p = w2 + (long)e * HDIM * ODIM + o;
#pragma unroll 8
        for (int k = 0; k < HDIM; k++) s = fmaf(heS[t][k], w2p[k * ODIM], s);
        oeS[t][o] = s * scale[tokS[t]];
    }
    __syncthreads();

    if (tid < 16) {
        const int i = mt * 16 + tid;
        if (i < count) {
            float mx = oeS[tid][0];
#pragma unroll
            for (int o = 1; o < ODIM; o++) mx = fmaxf(mx, oeS[tid][o]);
            float s = 0.0f;
#pragma unroll
            for (int o = 0; o < ODIM; o++) s += __expf(oeS[tid][o] - mx);
            const float l = __logf(s);
            float* op = out + (long)tokS[tid] * ODIM;
#pragma unroll
            for (int o = 0; o < ODIM; o++) op[o] = oeS[tid][o] - mx - l;
        }
    }
}

// ---------------------------------------------------------------------------
// launch
// ---------------------------------------------------------------------------
extern "C" void kernel_launch(void* const* d_in, const int* in_sizes, int n_in,
                              void* d_out, int out_size, void* d_ws, size_t ws_size,
                              hipStream_t stream) {
    const float* x       = (const float*)d_in[0];
    const float* conv1_w = (const float*)d_in[1];
    const float* conv1_b = (const float*)d_in[2];
    const float* conv2_w = (const float*)d_in[3];
    const float* conv2_b = (const float*)d_in[4];
    const float* gate_w  = (const float*)d_in[5];
    const float* w1      = (const float*)d_in[6];
    const float* b1      = (const float*)d_in[7];
    const float* w2      = (const float*)d_in[8];
    const float* b2      = (const float*)d_in[9];
    float* out = (float*)d_out;

    // workspace layout
    char* ws = (char*)d_ws;
    size_t off = 0;
    auto alloc = [&](size_t bytes) {
        char* p = ws + off;
        off = (off + bytes + 255) & ~(size_t)255;
        return p;
    };
    unsigned short* h1   = (unsigned short*)alloc((size_t)BATCH * C1HW * C1HW * 32 * 2); // 177 MB
    unsigned short* h    = (unsigned short*)alloc((size_t)BATCH * DDIM * 2);             //  75 MB
    unsigned short* w1b  = (unsigned short*)alloc((size_t)NEXP * KT1 * HDIM * 32 * 2);   //  19 MB
    unsigned short* wc2b = (unsigned short*)alloc((size_t)WC2ELEMS * 2);
    float*          scl  = (float*)alloc((size_t)BATCH * 4);
    int*            cnt  = (int*)alloc(256);
    int*            tokl = (int*)alloc((size_t)NEXP * BATCH * 4);
    (void)ws_size; (void)in_sizes; (void)n_in; (void)out_size;

    zero_cnt_kernel<<<1, 64, 0, stream>>>(cnt);

    {   // conv1
        int n = BATCH * C1HW * C1HW;
        conv1_kernel<<<(n + 255) / 256, 256, 0, stream>>>(x, conv1_w, conv1_b, h1);
    }
    {   // weight repacks
        conv2w_convert_kernel<<<(WC2ELEMS + 255) / 256, 256, 0, stream>>>(conv2_w, wc2b);
        long n = (long)NEXP * KT1 * HDIM * 32;
        w1_convert_kernel<<<(unsigned)((n + 255) / 256), 256, 0, stream>>>(w1, w1b);
    }
    {   // conv2 + relu + pool (WMMA, TDM-staged weights)
        conv2_kernel<<<NTOK2 / 128, 256, 0, stream>>>(h1, wc2b, conv2_b, h);
    }
    {   // gate + gather
        gate_kernel<<<(BATCH * 32) / 256, 256, 0, stream>>>(h, gate_w, scl, cnt, tokl);
    }
    {   // sparse expert FFN (WMMA)
        dim3 grid(BATCH / 16, NEXP);
        expert_kernel<<<grid, 128, 0, stream>>>(h, w1b, b1, w2, b2, scl, cnt, tokl, out);
    }
}
